// MetaConv_23313082483174
// MI455X (gfx1250) — matrix operand order repaired and checked by reading, not compiled, not run on hardware
//
#include <hip/hip_runtime.h>
#include <stdint.h>

typedef __attribute__((ext_vector_type(16))) __bf16 v16bf;
typedef __attribute__((ext_vector_type(8)))  float  v8f;

#define BS     32
#define EMB    512
#define GLAT   256
#define NOUT   64
#define NIN    64
#define NPOS   9
#define WFLAT  (NOUT*NIN*NPOS)    // 36864
#define H      96
#define WIDTH  96
#define HO     94
#define WO     94
#define TPR    6                  // pixel tiles per output row (ceil(94/16))
#define NTILE  (HO*TPR)           // 564 row-aligned pixel tiles

// ---------- kernel 1: hdn = relu(emb @ w1^T + b1)  [32,256] ----------
__global__ void k_hyper1(const float* __restrict__ emb, const float* __restrict__ w1,
                         const float* __restrict__ b1, float* __restrict__ hdn) {
  int idx = blockIdx.x * blockDim.x + threadIdx.x;      // 32*256 = 8192
  if (idx >= BS * GLAT) return;
  int b = idx / GLAT, j = idx % GLAT;
  const float* e = emb + (size_t)b * EMB;
  const float* w = w1  + (size_t)j * EMB;
  float s = b1[j];
  for (int l = 0; l < EMB; ++l) s += e[l] * w[l];
  hdn[idx] = s > 0.f ? s : 0.f;
}

// ---------- kernel 2: hypernet L2, scattering weights into WMMA A-fragment order ----
// Layout: wfrag[b][oc_tile(4)][pos(9)][ks(2)][lane(32)][t(16)]  (bf16)
// Contiguous k order (k = 16*hl + t), same as the B loads in k_conv_wmma:
//   lane = m + 16*hl,  hl = kk>>4,  t = kk&15
__global__ void k_hyper2(const float* __restrict__ hdn, const float* __restrict__ w2,
                         const float* __restrict__ b2, __bf16* __restrict__ wfrag) {
  size_t idx = (size_t)blockIdx.x * blockDim.x + threadIdx.x;   // 32*36864
  if (idx >= (size_t)BS * WFLAT) return;
  int b = (int)(idx / WFLAT), j = (int)(idx % WFLAT);
  int oc  = j / (NIN * NPOS);
  int rem = j % (NIN * NPOS);
  int ic  = rem / NPOS, pos = rem % NPOS;

  const float* hb = hdn + (size_t)b * GLAT;
  const float* w  = w2  + (size_t)j * GLAT;
  float s = b2[j];
  for (int l = 0; l < GLAT; ++l) s += hb[l] * w[l];

  int octile = oc >> 4, m = oc & 15;
  int ks = ic >> 5, kk = ic & 31;
  int hl = kk >> 4;
  int t  = kk & 15;
  int lane = m + 16 * hl;
  size_t off = (((((size_t)b * 4 + octile) * NPOS + pos) * 2 + ks) * 32 + lane) * 16 + t;
  wfrag[off] = (__bf16)s;
}

// ---------- kernel 3: x f32 NCHW -> bf16 channel-last [b][y][x][ic] ----------
// Channel-last makes every WMMA B-fragment (16 consecutive K=ic at one pixel)
// a single contiguous 32-byte per-lane load.
__global__ void k_cvtx(const float* __restrict__ x, __bf16* __restrict__ xt, size_t n) {
  size_t i = (size_t)blockIdx.x * blockDim.x + threadIdx.x;
  if (i >= n) return;
  // i indexes NCHW: b, ic, y, xcol
  int px   = (int)(i % (H * WIDTH));
  int bc   = (int)(i / (H * WIDTH));
  int ic   = bc % NIN;
  int b    = bc / NIN;
  xt[(((size_t)b * H * WIDTH) + px) * NIN + ic] = (__bf16)x[i];
}

// ---------- kernel 4: per-sample 3x3 conv, implicit GEMM via WMMA bf16 ----------
// grid: (141, 1, 32), block = 128 (4 wave32s). Each wave: one row-aligned 16-pixel
// tile x ALL 64 output channels (4 accumulators). NO LDS: A fragments come
// fragment-packed from k_hyper2; B fragments are direct v16bf loads from
// channel-last x. Inner loop is a pure b128-load + v_wmma stream.
__global__ __launch_bounds__(128) void k_conv_wmma(const __bf16* __restrict__ xt,
                                                   const v16bf* __restrict__ wfrag,
                                                   float* __restrict__ out) {
  const int wave = threadIdx.x >> 5;
  const int lane = threadIdx.x & 31;
  const int b    = blockIdx.z;

  const int tile = blockIdx.x * 4 + wave;     // 0..563
  const int oy   = tile / TPR;                // output row 0..93
  const int ox0  = (tile % TPR) * 16;         // 0,16,...,80

  const __bf16* xtb = xt + (size_t)b * H * WIDTH * NIN;
  const v16bf*  wf  = wfrag + (size_t)b * (4 * NPOS * 2 * 32);

  const int hl = lane >> 4;
  const int ml = lane & 15;

  v8f acc[4] = {v8f{}, v8f{}, v8f{}, v8f{}};

  for (int pos = 0; pos < NPOS; ++pos) {
    const int kh = pos / 3, kw = pos % 3;
    const int row = oy + kh;                  // 0..95
    int col = ox0 + ml + kw;                  // per-lane pixel column
    col = col < WIDTH ? col : WIDTH - 1;      // clamp (tail tiles; masked at store)
    const __bf16* bbase = xtb + ((size_t)row * WIDTH + col) * NIN + hl * 16;
    __builtin_prefetch(bbase + WIDTH * NIN, 0, 1);   // next kernel row
#pragma unroll
    for (int ks = 0; ks < 2; ++ks) {
      // --- B fragment: 16 consecutive ic at this pixel (32B aligned) ---
      v16bf bfrag = *(const v16bf*)(bbase + ks * 32);
      // --- 4 oc-tiles: A fragments direct (2x b128 each), 4 WMMAs ---
#pragma unroll
      for (int ot = 0; ot < 4; ++ot) {
        v16bf af = wf[(size_t)(((ot * NPOS + pos) * 2 + ks) * 32 + lane)];
        acc[ot] = __builtin_amdgcn_wmma_f32_16x16x32_bf16(
            false, af, false, bfrag, (short)0, acc[ot], false, false);
      }
    }
  }

  // --- store: f32 C/D layout: VGPR r -> M = r + 8*hl, N = ml ---
  const int ox = ox0 + ml;
  if (ox < WO) {
#pragma unroll
    for (int ot = 0; ot < 4; ++ot) {
#pragma unroll
      for (int r = 0; r < 8; ++r) {
        int oc = ot * 16 + r + 8 * hl;
        out[(((size_t)b * NOUT + oc) * HO + oy) * WO + ox] = acc[ot][r];
      }
    }
  }
}

extern "C" void kernel_launch(void* const* d_in, const int* in_sizes, int n_in,
                              void* d_out, int out_size, void* d_ws, size_t ws_size,
                              hipStream_t stream) {
  const float* emb  = (const float*)d_in[0];   // [32,512]
  const float* x    = (const float*)d_in[1];   // [32,64,96,96]
  const float* w1_w = (const float*)d_in[2];   // [256,512]
  const float* w1_b = (const float*)d_in[3];   // [256]
  const float* w2_w = (const float*)d_in[4];   // [36864,256]
  const float* w2_b = (const float*)d_in[5];   // [36864]
  float* out = (float*)d_out;                  // [32,64,94,94]

  char* ws = (char*)d_ws;
  float*  hdn   = (float*)(ws);                         // 32 KB
  __bf16* wfrag = (__bf16*)(ws + 32768);                // 32*36864*2 = 2.36 MB
  __bf16* xt    = (__bf16*)(ws + 32768 + 2359296);      // 37.75 MB (channel-last bf16)

  // 1) hdn = relu(emb @ w1^T + b1)
  k_hyper1<<<dim3((BS * GLAT + 255) / 256), dim3(256), 0, stream>>>(emb, w1_w, w1_b, hdn);

  // 2) per-sample weights, scattered into WMMA A-fragment order
  {
    size_t n = (size_t)BS * WFLAT;
    k_hyper2<<<dim3((unsigned)((n + 255) / 256)), dim3(256), 0, stream>>>(hdn, w2_w, w2_b, wfrag);
  }

  // 3) x -> bf16 channel-last transpose
  {
    size_t n = (size_t)BS * NIN * H * WIDTH;
    k_cvtx<<<dim3((unsigned)((n + 255) / 256)), dim3(256), 0, stream>>>(x, xt, n);
  }

  // 4) conv via WMMA: 564 row-aligned pixel tiles -> 141 blocks of 4 waves
  k_conv_wmma<<<dim3(141, 1, BS), dim3(128), 0, stream>>>(xt, (const v16bf*)wfrag, out);
}
